// SPGDeformableConv2d_56289841381836
// MI455X (gfx1250) — matrix operand order repaired and checked by reading, not compiled, not run on hardware
//
#include <hip/hip_runtime.h>

// SPG Deformable Conv2d for MI455X (gfx1250, wave32, WMMA).
// Dominant op: [40000 x 2304] x [2304 x 256] GEMM -> bf16 WMMA f32-accum.
//
// Workspace layout (bytes), total ~46.5 MB:
//   [0        ..    36)  kmask (9 floats)
//   [1024     .. +4.32M) conv outputs: offsets(18ch raw)+mask(9ch, post 2*sigmoid)
//   [4321024  .. +1.18M) packed masked weights bf16 [256][2304] (Kdim = c*9+k)
//   [5500672  .. +40.96M) xT: channel-last x [b][hw][c] f32

#define CIN   256
#define COUT  256
#define HH    25
#define WW    25
#define HW    625
#define KDIM  2304      // CIN*9
#define MTOT  40000     // B*HW
#define NSTEP 72        // KDIM/32

typedef __attribute__((ext_vector_type(8)))  float  v8f;
typedef __attribute__((ext_vector_type(16))) __bf16 v16bf;

__device__ __forceinline__ unsigned short f2bf(float f) {
  unsigned int u = __float_as_uint(f);
  u += 0x7FFFu + ((u >> 16) & 1u);          // round-to-nearest-even
  return (unsigned short)(u >> 16);
}

// ---------------------------------------------------------------- kmask ----
__global__ __launch_bounds__(64) void spg_kmask_kernel(
    const float* __restrict__ src, const int* __restrict__ seg,
    float* __restrict__ kmaskf) {
  __shared__ int s_ok[9];
  const int tid = threadIdx.x;              // tid == batch index (B == 64)
  if (tid < 9) s_ok[tid] = 1;
  __syncthreads();
  float code = src[((size_t)tid * (CIN + 1) + CIN) * HW + 12 * WW + 12];
  int ci = (int)code;
  int cx = ci / 1000, cy = ci - cx * 1000;
  int ctr = seg[cx * 100 + cy];
#pragma unroll
  for (int t = 0; t < 9; ++t) {
    int ok = (seg[(cx - 1 + t / 3) * 100 + (cy - 1 + t % 3)] == ctr) ? 1 : 0;
    atomicAnd(&s_ok[t], ok);
  }
  __syncthreads();
  if (tid < 9) kmaskf[tid] = s_ok[tid] ? 1.0f : 0.0f;
}

// ------------------------------------------------------- weight packing ----
__global__ __launch_bounds__(256) void spg_wpack_kernel(
    const float* __restrict__ regw, const float* __restrict__ kmaskf,
    unsigned short* __restrict__ wpack) {
  int t = blockIdx.x * 256 + threadIdx.x;   // grid = KDIM blocks -> 256*2304
  int n = t / KDIM;
  int kc = t - n * KDIM;
  int c = kc / 9;
  int k = kc - c * 9;
  float v = regw[((size_t)n * CIN + c) * 9 + k] * kmaskf[k];
  wpack[t] = f2bf(v);
}

// --------------------------------------------- offset/mod 3x3 conv (27ch) --
__global__ __launch_bounds__(256) void spg_conv_kernel(
    const float* __restrict__ src, const float* __restrict__ offw,
    const float* __restrict__ offb, const float* __restrict__ modw,
    const float* __restrict__ modb, float* __restrict__ offc) {
  __shared__ float s_wt[27 * 16 * 9];
  const int tid = threadIdx.x;
  const int pix = blockIdx.x * 256 + tid;
  const bool live = pix < MTOT;
  int b = 0, h = 0, w = 0, hw = 0;
  if (live) { b = pix / HW; hw = pix - b * HW; h = hw / WW; w = hw - h * WW; }
  float okf[9]; int o9[9];
#pragma unroll
  for (int t = 0; t < 9; ++t) {
    int hh = h + t / 3 - 1, w2 = w + t % 3 - 1;
    bool ok = (hh >= 0) && (hh < HH) && (w2 >= 0) && (w2 < WW);
    okf[t] = ok ? 1.0f : 0.0f;
    o9[t] = ok ? hh * WW + w2 : 0;
  }
  float acc[27];
#pragma unroll
  for (int ch = 0; ch < 27; ++ch) acc[ch] = 0.0f;

  for (int cc = 0; cc < CIN; cc += 16) {
    for (int idx = tid; idx < 27 * 16 * 9; idx += 256) {
      int ch = idx / 144, rem = idx - ch * 144;
      int ci2 = rem / 9, t9 = rem - ci2 * 9;
      int cin = cc + ci2;
      s_wt[idx] = (ch < 18) ? offw[((ch * CIN) + cin) * 9 + t9]
                            : modw[(((ch - 18) * CIN) + cin) * 9 + t9];
    }
    __syncthreads();
    if (live) {
#pragma unroll 1
      for (int ci2 = 0; ci2 < 16; ++ci2) {
        const float* xc = src + ((size_t)b * (CIN + 1) + (cc + ci2)) * HW;
        float xv[9];
#pragma unroll
        for (int t = 0; t < 9; ++t) xv[t] = okf[t] * xc[o9[t]];
        const float* wrow = s_wt + ci2 * 9;
#pragma unroll
        for (int ch = 0; ch < 27; ++ch) {
          float s = 0.0f;
#pragma unroll
          for (int t = 0; t < 9; ++t) s += wrow[ch * 144 + t] * xv[t];
          acc[ch] += s;
        }
      }
    }
    __syncthreads();
  }
  if (live) {
#pragma unroll
    for (int ch = 0; ch < 27; ++ch) {
      float v;
      if (ch < 18) v = acc[ch] + offb[ch];
      else { float t2 = acc[ch] + modb[ch - 18]; v = 2.0f / (1.0f + __expf(-t2)); }
      offc[((size_t)b * 27 + ch) * HW + hw] = v;
    }
  }
}

// ------------------------------------------------- channel-last transpose --
__global__ __launch_bounds__(256) void spg_xT_kernel(
    const float* __restrict__ src, float* __restrict__ xT) {
  int t = blockIdx.x * 256 + threadIdx.x;   // grid = 40000 -> 10.24M
  int c = t & 255;
  int pix = t >> 8;
  int b = pix / HW;
  int hw = pix - b * HW;
  xT[t] = src[((size_t)b * (CIN + 1) + c) * HW + hw];
}

// ------------------------------- fused bilinear-sample + bf16 WMMA GEMM ----
// Block: 128 threads (4 waves). One 16-row M-tile x all 256 N per block.
__global__ __launch_bounds__(128) void spg_gemm_kernel(
    const float* __restrict__ xT, const float* __restrict__ offc,
    const unsigned short* __restrict__ wpack, float* __restrict__ out) {
  __shared__ float          s_w[144 * 4];   // 4 corner weights (mask folded)
  __shared__ int            s_p[144 * 4];   // 4 corner offsets (*CIN)
  __shared__ int            s_rb[16];       // per-row xT batch base
  __shared__ unsigned short s_a[16 * 34];   // A tile bf16, padded stride

  const int tid = threadIdx.x;
  const int row0 = blockIdx.x * 16;

  if (tid < 144) {
    int m = tid / 9, kt = tid - m * 9;
    int row = row0 + m;
    int b = row / HW;
    int hw = row - b * HW;
    int h = hw / WW, w = hw - h * WW;
    const float* ob = offc + (size_t)b * 27 * HW + hw;
    float dy = ob[(kt * 2 + 0) * HW];
    float dx = ob[(kt * 2 + 1) * HW];
    float mk = ob[(18 + kt) * HW];
    float py = (float)h + (float)(kt / 3 - 1) + dy;
    float px = (float)w + (float)(kt % 3 - 1) + dx;
    float y0f = floorf(py), x0f = floorf(px);
    float wy1 = py - y0f, wx1 = px - x0f;
    int y0 = (int)y0f, x0 = (int)x0f;
#pragma unroll
    for (int cr = 0; cr < 4; ++cr) {
      int yi = y0 + (cr >> 1);
      int xi = x0 + (cr & 1);
      float wgt = ((cr >> 1) ? wy1 : 1.0f - wy1) *
                  ((cr & 1) ? wx1 : 1.0f - wx1) * mk;
      if (!((yi >= 0) && (yi < HH) && (xi >= 0) && (xi < WW))) wgt = 0.0f;
      int yc = yi < 0 ? 0 : (yi > HH - 1 ? HH - 1 : yi);
      int xc = xi < 0 ? 0 : (xi > WW - 1 ? WW - 1 : xi);
      s_w[tid * 4 + cr] = wgt;
      s_p[tid * 4 + cr] = (yc * WW + xc) * CIN;
    }
  }
  if (tid < 16) {
    int row = row0 + tid;
    int b = row / HW;
    s_rb[tid] = b * HW * CIN;
  }
  __syncthreads();

  v8f acc[4];
#pragma unroll
  for (int i = 0; i < 4; ++i)
#pragma unroll
    for (int r = 0; r < 8; ++r) acc[i][r] = 0.0f;

  const int lane = tid & 31;
  const int wv   = tid >> 5;
  const int half = lane >> 4;
  const int mrow = lane & 15;
  const int ncol = lane & 15;
  const unsigned int* s_a32 = reinterpret_cast<const unsigned int*>(s_a);

  for (int step = 0; step < NSTEP; ++step) {
    const int kkb = step * 32;
    // Cooperative A-tile build: 16 rows x 32 K (bilinear sample -> bf16).
#pragma unroll
    for (int q = 0; q < 4; ++q) {
      int e = tid + q * 128;
      int m = e & 15;
      int j = e >> 4;
      int kc = kkb + j;
      int c = kc / 9;
      int kt = kc - c * 9;
      int idx = (m * 9 + kt) * 4;
      int base = s_rb[m] + c;
      float v = s_w[idx + 0] * xT[base + s_p[idx + 0]] +
                s_w[idx + 1] * xT[base + s_p[idx + 1]] +
                s_w[idx + 2] * xT[base + s_p[idx + 2]] +
                s_w[idx + 3] * xT[base + s_p[idx + 3]];
      s_a[m * 34 + j] = f2bf(v);
    }
    __syncthreads();

    // A fragment, contiguous K order (matches B):
    // lanes 0-15: row m, K 0..15; lanes 16-31: row m, K 16..31.
    union { unsigned int u[8]; v16bf v; } afr;
#pragma unroll
    for (int r = 0; r < 8; ++r) {
      int K = 2 * r + 16 * half;
      afr.u[r] = s_a32[(mrow * 34 + K) >> 1];
    }

    // B fragments: lane = col n, 16 consecutive K (contiguous 32B in wpack).
    const int kb = kkb + 16 * half;
#pragma unroll
    for (int i = 0; i < 4; ++i) {
      int n = (wv * 4 + i) * 16 + ncol;
      const v16bf bfr =
          *reinterpret_cast<const v16bf*>(wpack + (size_t)n * KDIM + kb);
      acc[i] = __builtin_amdgcn_wmma_f32_16x16x32_bf16(
          false, afr.v, false, bfr, (short)0, acc[i], false, false);
    }
    if (step + 1 < NSTEP) {
      __builtin_prefetch(
          (const void*)(wpack + (size_t)((wv * 4) * 16 + ncol) * KDIM + kb + 32),
          0, 1);
    }
    __syncthreads();
  }

  // Epilogue: D layout -> lane holds col (lane&15), VGPR r -> row r+8*half.
#pragma unroll
  for (int i = 0; i < 4; ++i) {
    int n = (wv * 4 + i) * 16 + ncol;
#pragma unroll
    for (int r = 0; r < 8; ++r) {
      int row = row0 + r + 8 * half;
      int b = row / HW;
      int hw = row - b * HW;
      out[((size_t)b * COUT + n) * HW + hw] = acc[i][r];
    }
  }
}

// --------------------------------------------------------------- launch ----
extern "C" void kernel_launch(void* const* d_in, const int* in_sizes, int n_in,
                              void* d_out, int out_size, void* d_ws,
                              size_t ws_size, hipStream_t stream) {
  const float* src  = (const float*)d_in[0];  // source_x (B,257,25,25)
  const float* offw = (const float*)d_in[1];  // offset_w (18,256,3,3)
  const float* offb = (const float*)d_in[2];  // offset_b (18)
  const float* modw = (const float*)d_in[3];  // mod_w    (9,256,3,3)
  const float* modb = (const float*)d_in[4];  // mod_b    (9)
  const float* regw = (const float*)d_in[5];  // reg_w    (256,256,3,3)
  const int*   seg  = (const int*)d_in[6];    // segment  (100,100)
  float* out = (float*)d_out;

  char* ws = (char*)d_ws;
  float*          kmaskf = (float*)(ws);
  float*          offc   = (float*)(ws + 1024);
  unsigned short* wpack  = (unsigned short*)(ws + 1024 + 4320000);
  float*          xT     = (float*)(ws + 1024 + 4320000 + 1179648);

  spg_kmask_kernel<<<1, 64, 0, stream>>>(src, seg, kmaskf);
  spg_wpack_kernel<<<KDIM, 256, 0, stream>>>(regw, kmaskf, wpack);
  spg_conv_kernel<<<(MTOT + 255) / 256, 256, 0, stream>>>(src, offw, offb,
                                                          modw, modb, offc);
  spg_xT_kernel<<<40000, 256, 0, stream>>>(src, xT);
  spg_gemm_kernel<<<2500, 128, 0, stream>>>(xT, offc, wpack, out);
}
